// CPCLoss_5299989643840
// MI455X (gfx1250) — matrix operand order repaired and checked by reading, not compiled, not run on hardware
//
#include <hip/hip_runtime.h>
#include <hip/hip_bf16.h>

#define S_ 8
#define U_ 8
#define K_ 6
#define NNEG 17
#define ZD 64
#define CD 256
#define LEN 512
#define TT (LEN + K_)      // 518 time steps in z and c
#define LB 64              // l-rows per workgroup
#define NTHREADS 128       // 4 wave32 waves

#define WS_WBF_BYTE_OFF 4096                      // bf16 W image starts here in ws
#define WBLK (ZD * CD)                            // 16384 elements per k
#define WS_NEEDED (WS_WBF_BYTE_OFF + K_ * WBLK * 2)

typedef __attribute__((ext_vector_type(16))) __bf16 v16bf;
typedef __attribute__((ext_vector_type(8)))  __bf16 v8bf;
typedef __attribute__((ext_vector_type(8)))  float  v8f;
typedef __attribute__((ext_vector_type(4)))  unsigned int v4u;
typedef __attribute__((ext_vector_type(8)))  int v8i;
typedef __attribute__((ext_vector_type(4)))  int v4i;

static __device__ __forceinline__ __bf16 f2bf(float f) {
  // round-to-nearest-even f32 -> bf16
  unsigned u = __builtin_bit_cast(unsigned, f);
  unsigned r = (u + 0x7FFFu + ((u >> 16) & 1u)) >> 16;
  unsigned short hs = (unsigned short)r;
  return __builtin_bit_cast(__bf16, hs);
}

// ---- one-time: W (f32) -> bf16 in WMMA-B fragment layout [k][c/16][z][c%16]
__global__ void cpc_prep(const float* __restrict__ Wg,
                         unsigned short* __restrict__ wbf) {
  int i = blockIdx.x * blockDim.x + threadIdx.x;
  if (i >= K_ * WBLK) return;
  int k   = i / WBLK;
  int rem = i - k * WBLK;
  int z   = rem >> 8;       // 0..63
  int cc  = rem & 255;      // 0..255
  __bf16 v = f2bf(Wg[((size_t)k * ZD + z) * CD + cc]);
  wbf[(size_t)k * WBLK + (((cc >> 4) * ZD) + z) * 16 + (cc & 15)] =
      __builtin_bit_cast(unsigned short, v);
}

__global__ __launch_bounds__(NTHREADS) void cpc_main(
    const float* __restrict__ zg, const float* __restrict__ cg,
    const float* __restrict__ Wg, const float* __restrict__ bg,
    const int* __restrict__ bidx, const int* __restrict__ sidx,
    float* __restrict__ ws, const unsigned short* __restrict__ wbf,
    int useTdm)
{
  // LDS: W[k]^T in WMMA-B-fragment layout [c/16][z][c%16] (bf16),
  // plus bias, the f32 Wc tile, the f logits buffer, and reduction scratch.
  __shared__ __bf16 WB[WBLK];
  __shared__ float  bArr[ZD];
  __shared__ float  WcS[LB * ZD];
  __shared__ float  fBuf[LB * (NNEG + 1)];
  __shared__ float  red[2 * NTHREADS];

  const int tid  = threadIdx.x;
  const int wv   = tid >> 5;          // wave id 0..3 (wave32)
  const int lane = tid & 31;
  const int mrow = lane & 15;
  const int hiH  = (lane >> 4) & 1;   // lanes 16..31

  const int wg = blockIdx.x;          // 0..511
  const int su = wg >> 3;             // 0..63  == s*8+u
  const int lb = wg & 7;
  const int s  = su >> 3;
  const int u  = su & 7;
  const int l0 = lb * LB;

  // ---- Load A fragments (c tile rows, f32->bf16). k-independent: load once,
  //      reuse across all 6 k iterations. Layout (contiguous, matches B):
  //      elems 0..15 -> K = 32*kk + 16*hiH + 0..15.
  v16bf aFrag[8];
  {
    const float* crow = cg + ((size_t)su * TT + (size_t)(l0 + wv * 16 + mrow)) * CD;
    #pragma unroll
    for (int kk = 0; kk < 8; ++kk) {
      const int base0 = kk * 32 + hiH * 16;
      const int base1 = base0 + 8;
      float4 a0 = *(const float4*)(crow + base0);
      float4 a1 = *(const float4*)(crow + base0 + 4);
      float4 a2 = *(const float4*)(crow + base1);
      float4 a3 = *(const float4*)(crow + base1 + 4);
      v16bf a;
      a[0]  = f2bf(a0.x); a[1]  = f2bf(a0.y); a[2]  = f2bf(a0.z); a[3]  = f2bf(a0.w);
      a[4]  = f2bf(a1.x); a[5]  = f2bf(a1.y); a[6]  = f2bf(a1.z); a[7]  = f2bf(a1.w);
      a[8]  = f2bf(a2.x); a[9]  = f2bf(a2.y); a[10] = f2bf(a2.z); a[11] = f2bf(a2.w);
      a[12] = f2bf(a3.x); a[13] = f2bf(a3.y); a[14] = f2bf(a3.z); a[15] = f2bf(a3.w);
      aFrag[kk] = a;
    }
  }

  const float scale = 0.125f;  // 1/sqrt(ZD)
  const unsigned ldsWB = (unsigned)(size_t)(void*)&WB[0];   // LDS byte offset
  const unsigned long long wbfBase = (unsigned long long)(size_t)wbf;

  for (int k = 0; k < K_; ++k) {
    __syncthreads();  // previous iteration's readers of WB/WcS/fBuf are done

    // ---- stage W[k] fragments into LDS ----
    if (useTdm) {
      if (wv == 0) {
        // Tensor Data Mover: 1-D copy of 32KB (16384 x 2B) L2 -> LDS.
        // D# per CDNA5 ISA.
        unsigned long long ga = wbfBase + (unsigned long long)k * (WBLK * 2);
        v4u g0;
        g0[0] = 1u;                                   // count=1 (valid user D#)
        g0[1] = ldsWB;                                // lds_addr [63:32]
        g0[2] = (unsigned)(ga & 0xFFFFFFFFu);         // global_addr lo
        g0[3] = (unsigned)((ga >> 32) & 0x01FFFFFFu)  // global_addr [56:32]
                | (2u << 30);                         // type=2 ("image")
        v8i g1;
        g1[0] = (int)(1u << 16);                      // data_size=1 -> 2 bytes
        g1[1] = (int)((WBLK & 0xFFFFu) << 16);        // tensor_dim0 lo16
        g1[2] = (int)((WBLK >> 16) | (1u << 16));     // tensor_dim0 hi16 | tensor_dim1=1
        g1[3] = (int)((WBLK & 0xFFFFu) << 16);        // tile_dim0 = 16384
        g1[4] = 1;                                    // tile_dim1 = 1
        g1[5] = WBLK;                                 // tensor_dim0_stride lo32
        g1[6] = 0;
        g1[7] = 0;
        v4i g2 = {0, 0, 0, 0};
        v4i g3 = {0, 0, 0, 0};
        v8i g4 = {0, 0, 0, 0, 0, 0, 0, 0};            // unused extra group
        __builtin_amdgcn_tensor_load_to_lds(g0, g1, g2, g3, g4, 0);
        __builtin_amdgcn_s_wait_tensorcnt(0);
      }
    } else {
      // fallback: convert f32 W[k] -> bf16 fragments in LDS directly
      for (int i = tid; i < WBLK; i += NTHREADS) {
        int z  = i >> 8;
        int cc = i & 255;
        WB[(((cc >> 4) * ZD) + z) * 16 + (cc & 15)] =
            f2bf(Wg[((size_t)k * ZD + z) * CD + cc]);
      }
    }
    if (tid < ZD) bArr[tid] = bg[k * ZD + tid];
    __syncthreads();

    // ---- WMMA GEMM: Wc[l, z] = sum_c c_s[l,c] * W[k][z,c]  (M=64,N=64,K=256)
    v8f acc[4] = {};
    #pragma unroll
    for (int kk = 0; kk < 8; ++kk) {
      v16bf a = aFrag[kk];
      const int g = kk * 2 + hiH;   // c-group of 16 for this half-wave
      #pragma unroll
      for (int nt = 0; nt < 4; ++nt) {
        const __bf16* bp = &WB[((g * ZD) + (nt * 16 + mrow)) * 16];
        v8bf blo = *(const v8bf*)bp;
        v8bf bhi = *(const v8bf*)(bp + 8);
        v16bf bfr;
        #pragma unroll
        for (int i2 = 0; i2 < 8; ++i2) { bfr[i2] = blo[i2]; bfr[i2 + 8] = bhi[i2]; }
        acc[nt] = __builtin_amdgcn_wmma_f32_16x16x32_bf16(
            false, a, false, bfr, (short)0, acc[nt], false, false);
      }
    }
    // C/D layout: VGPR r -> M = r + 8*hiH, N = lane%16. Add bias, store to LDS.
    #pragma unroll
    for (int nt = 0; nt < 4; ++nt) {
      const int col = nt * 16 + mrow;
      const float bv = bArr[col];
      #pragma unroll
      for (int r = 0; r < 8; ++r) {
        const int row = wv * 16 + hiH * 8 + r;
        WcS[row * ZD + col] = acc[nt][r] + bv;
      }
    }
    __syncthreads();

    // ---- dot phase: f[l][0] = pos, f[l][1+n] = negatives (gather from L2) ----
    for (int idx = tid; idx < LB * (NNEG + 1); idx += NTHREADS) {
      const int ll = idx / (NNEG + 1);
      const int j  = idx - ll * (NNEG + 1);
      const int l  = l0 + ll;
      const float* zrow;
      if (j == 0) {
        zrow = zg + ((size_t)su * TT + (size_t)(k + 1 + l)) * ZD;
      } else {
        const int n  = j - 1;
        const int bi = bidx[(k * U_ + u) * NNEG + n];
        const int si = sidx[(((((size_t)k * S_ + s) * U_ + u) * NNEG) + n) * LEN + l];
        zrow = zg + ((size_t)(s * U_ + bi) * TT + (size_t)(k + 1 + si)) * ZD;
      }
      const float* wr = &WcS[ll * ZD];
      float d = 0.f;
      #pragma unroll
      for (int q = 0; q < 16; ++q) {
        float4 zv = ((const float4*)zrow)[q];
        d += zv.x * wr[4 * q] + zv.y * wr[4 * q + 1] +
             zv.z * wr[4 * q + 2] + zv.w * wr[4 * q + 3];
      }
      fBuf[ll * (NNEG + 1) + j] = d * scale;
    }
    __syncthreads();

    // ---- per-l logsumexp over 18 candidates; loss and accuracy ----
    float lossLoc = 0.f, accLoc = 0.f;
    for (int ll = tid; ll < LB; ll += NTHREADS) {
      const float* fr = &fBuf[ll * (NNEG + 1)];
      const float f0 = fr[0];
      float mneg = -1e30f;
      #pragma unroll
      for (int j = 1; j < NNEG + 1; ++j) mneg = fmaxf(mneg, fr[j]);
      const float m = fmaxf(f0, mneg);
      float ssum = 0.f;
      #pragma unroll
      for (int j = 0; j < NNEG + 1; ++j) ssum += __expf(fr[j] - m);
      lossLoc += (m + __logf(ssum)) - f0;
      accLoc  += (f0 >= mneg) ? 1.f : 0.f;
    }

    // ---- block reduction, then atomics into workspace accumulators ----
    red[tid] = lossLoc;
    red[NTHREADS + tid] = accLoc;
    __syncthreads();
    for (int off = NTHREADS / 2; off > 0; off >>= 1) {
      if (tid < off) {
        red[tid] += red[tid + off];
        red[NTHREADS + tid] += red[NTHREADS + tid + off];
      }
      __syncthreads();
    }
    if (tid == 0) {
      atomicAdd(&ws[0], red[0]);            // global loss sum (all k)
      atomicAdd(&ws[1 + k], red[NTHREADS]); // per-k accuracy count
    }
  }
}

__global__ void cpc_init(float* __restrict__ ws) {
  if (threadIdx.x < 1 + K_) ws[threadIdx.x] = 0.f;
}

__global__ void cpc_fin(const float* __restrict__ ws, float* __restrict__ out) {
  const int t = threadIdx.x;
  if (t == 0)               out[0] = ws[0] * (1.0f / (float)(K_ * S_ * U_ * LEN));
  else if (t < 1 + K_)      out[t] = ws[t] * (1.0f / (float)(S_ * U_ * LEN));
}

extern "C" void kernel_launch(void* const* d_in, const int* in_sizes, int n_in,
                              void* d_out, int out_size, void* d_ws, size_t ws_size,
                              hipStream_t stream) {
  (void)in_sizes; (void)n_in; (void)out_size;
  const float* z    = (const float*)d_in[0];
  const float* c    = (const float*)d_in[1];
  const float* W    = (const float*)d_in[2];
  const float* b    = (const float*)d_in[3];
  const int*   bidx = (const int*)d_in[4];
  const int*   sidx = (const int*)d_in[5];
  float* out = (float*)d_out;
  float* ws  = (float*)d_ws;
  unsigned short* wbf =
      (unsigned short*)((char*)d_ws + WS_WBF_BYTE_OFF);
  const int useTdm = (ws_size >= (size_t)WS_NEEDED) ? 1 : 0;

  hipLaunchKernelGGL(cpc_init, dim3(1), dim3(32), 0, stream, ws);
  if (useTdm) {
    hipLaunchKernelGGL(cpc_prep, dim3((K_ * WBLK + 255) / 256), dim3(256), 0,
                       stream, W, wbf);
  }
  hipLaunchKernelGGL(cpc_main, dim3(S_ * U_ * (LEN / LB)), dim3(NTHREADS), 0, stream,
                     z, c, W, b, bidx, sidx, ws, wbf, useTdm);
  hipLaunchKernelGGL(cpc_fin, dim3(1), dim3(32), 0, stream, ws, out);
}
